// HyperSAGE_11639361372220
// MI455X (gfx1250) — compile-verified, matrix operand order checked by reading, not run on hardware
//
#include <hip/hip_runtime.h>
#include <math.h>

typedef __attribute__((ext_vector_type(16))) __bf16 v16bf;
typedef __attribute__((ext_vector_type(8)))  __bf16 v8bf;
typedef __attribute__((ext_vector_type(8)))  float  v8f;

constexpr int N_NODES = 50000;
constexpr int M_HE    = 5000;
constexpr int E_EDGES = 800000;
constexpr int N_INC   = 400000;
constexpr int FIN     = 128;
constexpr int HID     = 256;
constexpr int NC      = 40;
constexpr int NCP     = 48;   // padded output rows for O=40 weights (3x16)
constexpr int KHP     = 64;   // padded K for the head GEMM (40 -> 64)

// ---------------------------------------------------------------- utilities
__global__ void fill_zero_k(float* __restrict__ p, int n) {
  int i = blockIdx.x * blockDim.x + threadIdx.x;
  if (i < n) p[i] = 0.0f;
}

__global__ void count1_k(const int* __restrict__ idx, float* __restrict__ cnt, int n) {
  int i = blockIdx.x * blockDim.x + threadIdx.x;
  if (i < n) atomicAdd(cnt + idx[i], 1.0f);
}

__global__ void count2_k(const int* __restrict__ ia, float* __restrict__ ca,
                         const int* __restrict__ ib, float* __restrict__ cb, int n) {
  int i = blockIdx.x * blockDim.x + threadIdx.x;
  if (i < n) {
    atomicAdd(ca + ia[i], 1.0f);
    atomicAdd(cb + ib[i], 1.0f);
  }
}

// mode 0: d = 1/max(d,1)   mode 1: d = (d>0) ? 1/d : 0
__global__ void recip_k(float* __restrict__ d, int n, int mode) {
  int i = blockIdx.x * blockDim.x + threadIdx.x;
  if (i >= n) return;
  float v = d[i];
  if (mode == 0) d[i] = 1.0f / fmaxf(v, 1.0f);
  else           d[i] = (v > 0.0f) ? (1.0f / v) : 0.0f;
}

// Wp[Opad x Kpad] (bf16, row-major) = zero-padded convert of W[O x K] (stride ldw)
__global__ void pack_w_bf16_k(const float* __restrict__ W, int O, int K, int ldw,
                              __bf16* __restrict__ Wp, int Opad, int Kpad) {
  int gid = blockIdx.x * blockDim.x + threadIdx.x;
  int total = Opad * Kpad;
  if (gid >= total) return;
  int o = gid / Kpad, k = gid - o * Kpad;
  Wp[gid] = (o < O && k < K) ? (__bf16)W[(size_t)o * ldw + k] : (__bf16)0.0f;
}

// out[to[e]*ldo + f] += feat[from[e]*F + f] * (escale ? escale[from[e]] : 1)
__global__ void scatter_k(const float* __restrict__ feat,
                          const int* __restrict__ from, const int* __restrict__ to,
                          const float* __restrict__ escale,
                          float* __restrict__ out, int nnz, int F, int fshift, int ldo) {
  unsigned gid = blockIdx.x * blockDim.x + threadIdx.x;
  unsigned total = (unsigned)nnz * (unsigned)F;
  if (gid >= total) return;
  unsigned e, f;
  if (fshift >= 0) { e = gid >> fshift; f = gid & ((unsigned)F - 1u); }
  else             { e = gid / (unsigned)F; f = gid - e * (unsigned)F; }
  int fi = from[e];
  float v = feat[(size_t)fi * F + f];
  if (escale) v *= escale[fi];
  atomicAdd(out + (size_t)to[e] * ldo + f, v);
}

// buf[r*ldo+f] = relu?( buf[r*ldo+f]*dinv[r] + b[f] )
__global__ void epi_k(float* __restrict__ buf, const float* __restrict__ dinv,
                      const float* __restrict__ bias, int rows, int F, int fshift,
                      int ldo, int relu) {
  unsigned gid = blockIdx.x * blockDim.x + threadIdx.x;
  unsigned total = (unsigned)rows * (unsigned)F;
  if (gid >= total) return;
  unsigned r, f;
  if (fshift >= 0) { r = gid >> fshift; f = gid & ((unsigned)F - 1u); }
  else             { r = gid / (unsigned)F; f = gid - r * (unsigned)F; }
  size_t a = (size_t)r * ldo + f;
  float v = buf[a] * dinv[r] + bias[f];
  if (relu) v = fmaxf(v, 0.0f);
  buf[a] = v;
}

__global__ void log_softmax_k(float* __restrict__ out, int n, int C) {
  int i = blockIdx.x * blockDim.x + threadIdx.x;
  if (i >= n) return;
  float* row = out + (size_t)i * C;
  float m = -INFINITY;
  for (int c = 0; c < C; ++c) m = fmaxf(m, row[c]);
  float s = 0.0f;
  for (int c = 0; c < C; ++c) s += __expf(row[c] - m);
  float l = __logf(s) + m;
  for (int c = 0; c < C; ++c) row[c] = row[c] - l;
}

// ---------------------------------------------------------------- WMMA GEMM
// 16-bit A 16x32 / B 32x16 tile: lane&15 = M row (A) or N col (B);
// lane>>4 selects the K-interleave half:
//   elems [0..7]  -> K = k0 + lh*8 + j
//   elems [8..15] -> K = k0 + 16 + lh*8 + j
// A: fp32 in global, unconditional b128 loads + cvt_pk (amortized over OT wmmas).
// B: pre-converted bf16 weights, two unconditional b128 loads, no cvt,
//    software-pipelined one tile ahead so wmma overlaps B-tile L2 latency.
template <bool SCALE>
__device__ __forceinline__ v16bf load_a_tile(const float* __restrict__ row,
                                             int k0, int lh, float s) {
  const float4* rp = (const float4*)row;
  int i1 = (k0 + lh * 8) >> 2;
  int i2 = (k0 + 16 + lh * 8) >> 2;
  float4 p0 = rp[i1], p1 = rp[i1 + 1];
  float4 q0 = rp[i2], q1 = rp[i2 + 1];
  v16bf a;
  if (SCALE) {
    a[0] = (__bf16)(p0.x * s);  a[1] = (__bf16)(p0.y * s);
    a[2] = (__bf16)(p0.z * s);  a[3] = (__bf16)(p0.w * s);
    a[4] = (__bf16)(p1.x * s);  a[5] = (__bf16)(p1.y * s);
    a[6] = (__bf16)(p1.z * s);  a[7] = (__bf16)(p1.w * s);
    a[8] = (__bf16)(q0.x * s);  a[9] = (__bf16)(q0.y * s);
    a[10] = (__bf16)(q0.z * s); a[11] = (__bf16)(q0.w * s);
    a[12] = (__bf16)(q1.x * s); a[13] = (__bf16)(q1.y * s);
    a[14] = (__bf16)(q1.z * s); a[15] = (__bf16)(q1.w * s);
  } else {
    a[0] = (__bf16)p0.x;  a[1] = (__bf16)p0.y;
    a[2] = (__bf16)p0.z;  a[3] = (__bf16)p0.w;
    a[4] = (__bf16)p1.x;  a[5] = (__bf16)p1.y;
    a[6] = (__bf16)p1.z;  a[7] = (__bf16)p1.w;
    a[8] = (__bf16)q0.x;  a[9] = (__bf16)q0.y;
    a[10] = (__bf16)q0.z; a[11] = (__bf16)q0.w;
    a[12] = (__bf16)q1.x; a[13] = (__bf16)q1.y;
    a[14] = (__bf16)q1.z; a[15] = (__bf16)q1.w;
  }
  return a;
}

__device__ __forceinline__ v16bf load_b_tile(const __bf16* __restrict__ row,
                                             int k0, int lh) {
  const v8bf* rp = (const v8bf*)row;             // 16B granules
  v8bf lo = rp[(k0 + lh * 8) >> 3];
  v8bf hi = rp[(k0 + 16 + lh * 8) >> 3];
  return __builtin_shufflevector(lo, hi, 0, 1, 2, 3, 4, 5, 6, 7,
                                 8, 9, 10, 11, 12, 13, 14, 15);
}

// out[nrows x O (stride ldo)] = relu( A1*diag(scale1) @ W1^T (+ A2 @ W2^T) + bias )
// One wave per 16-row tile; full output strip (OT 16-col tiles) in VGPR
// accumulators: A streams from HBM exactly once, bf16 W rereads hit L2.
template <int OT, bool DUAL, bool SCALE>
__global__ __launch_bounds__(256)
void wmma_gemm_k(const float* __restrict__ A1, const float* __restrict__ scale1,
                 const __bf16* __restrict__ W1, int ldw1,
                 const float* __restrict__ A2,
                 const __bf16* __restrict__ W2, int ldw2,
                 const float* __restrict__ bias, float* __restrict__ out,
                 int nrows, int K, int O, int ldo, int relu) {
  const int lane = threadIdx.x & 31;
  const int wave = threadIdx.x >> 5;
  const int rowTile = blockIdx.x * (blockDim.x >> 5) + wave;
  const int r0 = rowTile * 16;
  if (r0 >= nrows) return;                 // wave-uniform: EXEC stays all-ones
  const int lr = lane & 15;
  const int lh = lane >> 4;

  const float s1 = SCALE ? scale1[r0 + lr] : 1.0f;
  const float* a1row = A1 + (size_t)(r0 + lr) * K;
  const float* a2row = DUAL ? (A2 + (size_t)(r0 + lr) * K) : nullptr;
  const __bf16* w1row = W1 + (size_t)lr * ldw1;   // per-lane base of B column lr
  const __bf16* w2row = DUAL ? (W2 + (size_t)lr * ldw2) : nullptr;

  v8f acc[OT];
#pragma unroll
  for (int ot = 0; ot < OT; ++ot)
    acc[ot] = (v8f){0.f, 0.f, 0.f, 0.f, 0.f, 0.f, 0.f, 0.f};

  for (int k0 = 0; k0 < K; k0 += 32) {
    v16bf a1v = load_a_tile<SCALE>(a1row, k0, lh, s1);
    v16bf a2v;
    if (DUAL) a2v = load_a_tile<false>(a2row, k0, lh, 1.0f);

    // software pipeline: B tiles for ot+1 are issued before the wmmas for ot,
    // so the wait before each wmma leaves the next tile's loads in flight.
    v16bf c1 = load_b_tile(w1row, k0, lh);
    v16bf c2;
    if (DUAL) c2 = load_b_tile(w2row, k0, lh);
#pragma unroll
    for (int ot = 0; ot < OT; ++ot) {
      v16bf n1, n2;
      if (ot + 1 < OT) {
        n1 = load_b_tile(w1row + (size_t)(ot + 1) * 16 * ldw1, k0, lh);
        if (DUAL) n2 = load_b_tile(w2row + (size_t)(ot + 1) * 16 * ldw2, k0, lh);
      }
      acc[ot] = __builtin_amdgcn_wmma_f32_16x16x32_bf16(
          false, a1v, false, c1, (short)0, acc[ot], false, false);
      if (DUAL)
        acc[ot] = __builtin_amdgcn_wmma_f32_16x16x32_bf16(
            false, a2v, false, c2, (short)0, acc[ot], false, false);
      if (ot + 1 < OT) {
        c1 = n1;
        if (DUAL) c2 = n2;
      }
    }
  }

  // D layout: VGPR v, lanes 0-15 -> M=v; lanes 16-31 -> M=v+8; N = lane&15.
#pragma unroll
  for (int ot = 0; ot < OT; ++ot) {
    int n = ot * 16 + lr;
    if (n < O) {
      float bv = bias ? bias[n] : 0.0f;
#pragma unroll
      for (int v = 0; v < 8; ++v) {
        int m = r0 + lh * 8 + v;
        float val = acc[ot][v] + bv;
        if (relu) val = fmaxf(val, 0.0f);
        out[(size_t)m * ldo + n] = val;
      }
    }
  }
}

// ---------------------------------------------------------------- host side
static inline int cdiv(long long a, long long b) { return (int)((a + b - 1) / b); }

extern "C" void kernel_launch(void* const* d_in, const int* in_sizes, int n_in,
                              void* d_out, int out_size, void* d_ws, size_t ws_size,
                              hipStream_t stream) {
  const float* x       = (const float*)d_in[0];
  const int*   src     = (const int*)d_in[1];
  const int*   dst     = (const int*)d_in[2];
  const int*   he_node = (const int*)d_in[3];
  const int*   he_edge = (const int*)d_in[4];
  const float* s_wl[3] = {(const float*)d_in[5], (const float*)d_in[8],  (const float*)d_in[11]};
  const float* s_wr[3] = {(const float*)d_in[6], (const float*)d_in[9],  (const float*)d_in[12]};
  const float* s_b[3]  = {(const float*)d_in[7], (const float*)d_in[10], (const float*)d_in[13]};
  const float* h_w[5]  = {(const float*)d_in[14], (const float*)d_in[16], (const float*)d_in[18],
                          (const float*)d_in[20], (const float*)d_in[22]};
  const float* h_b[5]  = {(const float*)d_in[15], (const float*)d_in[17], (const float*)d_in[19],
                          (const float*)d_in[21], (const float*)d_in[23]};
  const float* lp_w = (const float*)d_in[24];
  const float* lp_b = (const float*)d_in[25];
  float* out = (float*)d_out;

  float* ws    = (float*)d_ws;
  float* H0    = ws;                                  // [N,256]
  float* H1    = H0 + (size_t)N_NODES * HID;          // [N,256]
  float* XT    = H1 + (size_t)N_NODES * HID;          // [N,256] agg / xt scratch
  float* EF    = XT + (size_t)N_NODES * HID;          // [M,256]
  float* XS    = EF + (size_t)M_HE * HID;             // [N,64] sage out (padded)
  float* XH    = XS + (size_t)N_NODES * KHP;          // [N,64] hyper out (padded)
  float* RDEG  = XH + (size_t)N_NODES * KHP;          // [N]
  float* BINV  = RDEG + N_NODES;                      // [M]
  float* DINV  = BINV + M_HE;                         // [N]

  // bf16 weight arena (16B aligned: float offset above is a multiple of 4)
  __bf16* WB = (__bf16*)(DINV + N_NODES);
  __bf16* B_S0WL = WB;                    // [256,128]
  __bf16* B_S0WR = B_S0WL + HID * FIN;    // [256,128]
  __bf16* B_S1WL = B_S0WR + HID * FIN;    // [256,256]
  __bf16* B_S1WR = B_S1WL + HID * HID;    // [256,256]
  __bf16* B_S2WL = B_S1WR + HID * HID;    // [48,256]
  __bf16* B_S2WR = B_S2WL + NCP * HID;    // [48,256]
  __bf16* B_H0   = B_S2WR + NCP * HID;    // [256,128]
  __bf16* B_H1   = B_H0 + HID * FIN;      // [256,256]
  __bf16* B_H2   = B_H1 + HID * HID;      // [256,256]
  __bf16* B_H3   = B_H2 + HID * HID;      // [256,256]
  __bf16* B_H4   = B_H3 + HID * HID;      // [48,256]
  __bf16* B_LW1  = B_H4 + NCP * HID;      // [48,64]
  __bf16* B_LW2  = B_LW1 + NCP * KHP;     // [48,64]

  const int T = 256;
  const int gemmGrid = cdiv(N_NODES / 16, 8);         // 8 waves per block

  // --- degrees / inverse degrees (once, reused by all layers) ---
  fill_zero_k<<<cdiv(N_NODES, T), T, 0, stream>>>(RDEG, N_NODES);
  fill_zero_k<<<cdiv(M_HE, T),    T, 0, stream>>>(BINV, M_HE);
  fill_zero_k<<<cdiv(N_NODES, T), T, 0, stream>>>(DINV, N_NODES);
  count1_k<<<cdiv(E_EDGES, T), T, 0, stream>>>(dst, RDEG, E_EDGES);
  count2_k<<<cdiv(N_INC, T),   T, 0, stream>>>(he_edge, BINV, he_node, DINV, N_INC);
  recip_k<<<cdiv(N_NODES, T), T, 0, stream>>>(RDEG, N_NODES, 0);
  recip_k<<<cdiv(M_HE, T),    T, 0, stream>>>(BINV, M_HE, 1);
  recip_k<<<cdiv(N_NODES, T), T, 0, stream>>>(DINV, N_NODES, 1);

  // --- one-time bf16 weight conversion (zero-padded where O=40 / K=40) ---
  pack_w_bf16_k<<<cdiv(HID * FIN, T), T, 0, stream>>>(s_wl[0], HID, FIN, FIN, B_S0WL, HID, FIN);
  pack_w_bf16_k<<<cdiv(HID * FIN, T), T, 0, stream>>>(s_wr[0], HID, FIN, FIN, B_S0WR, HID, FIN);
  pack_w_bf16_k<<<cdiv(HID * HID, T), T, 0, stream>>>(s_wl[1], HID, HID, HID, B_S1WL, HID, HID);
  pack_w_bf16_k<<<cdiv(HID * HID, T), T, 0, stream>>>(s_wr[1], HID, HID, HID, B_S1WR, HID, HID);
  pack_w_bf16_k<<<cdiv(NCP * HID, T), T, 0, stream>>>(s_wl[2], NC, HID, HID, B_S2WL, NCP, HID);
  pack_w_bf16_k<<<cdiv(NCP * HID, T), T, 0, stream>>>(s_wr[2], NC, HID, HID, B_S2WR, NCP, HID);
  pack_w_bf16_k<<<cdiv(HID * FIN, T), T, 0, stream>>>(h_w[0], HID, FIN, FIN, B_H0, HID, FIN);
  pack_w_bf16_k<<<cdiv(HID * HID, T), T, 0, stream>>>(h_w[1], HID, HID, HID, B_H1, HID, HID);
  pack_w_bf16_k<<<cdiv(HID * HID, T), T, 0, stream>>>(h_w[2], HID, HID, HID, B_H2, HID, HID);
  pack_w_bf16_k<<<cdiv(HID * HID, T), T, 0, stream>>>(h_w[3], HID, HID, HID, B_H3, HID, HID);
  pack_w_bf16_k<<<cdiv(NCP * HID, T), T, 0, stream>>>(h_w[4], NC, HID, HID, B_H4, NCP, HID);
  pack_w_bf16_k<<<cdiv(NCP * KHP, T), T, 0, stream>>>(lp_w,      NC, NC, 2 * NC, B_LW1, NCP, KHP);
  pack_w_bf16_k<<<cdiv(NCP * KHP, T), T, 0, stream>>>(lp_w + NC, NC, NC, 2 * NC, B_LW2, NCP, KHP);

  // padded head inputs: zero so pad columns contribute nothing to head GEMM
  fill_zero_k<<<cdiv((long long)N_NODES * KHP, T), T, 0, stream>>>(XS, N_NODES * KHP);
  fill_zero_k<<<cdiv((long long)N_NODES * KHP, T), T, 0, stream>>>(XH, N_NODES * KHP);

  // --- SAGE branch: agg = scatter(x[src] -> dst); h = relu(agg/deg@wl + x@wr + b)
  const float* sin[3] = {x, H0, H1};
  int sI[3] = {FIN, HID, HID};
  for (int li = 0; li < 3; ++li) {
    const float* xin = sin[li];
    int I = sI[li];
    int ish = (I == 128) ? 7 : 8;
    fill_zero_k<<<cdiv((long long)N_NODES * I, T), T, 0, stream>>>(XT, N_NODES * I);
    scatter_k<<<cdiv((long long)E_EDGES * I, T), T, 0, stream>>>(
        xin, src, dst, nullptr, XT, E_EDGES, I, ish, I);
    if (li == 0)
      wmma_gemm_k<16, true, true><<<gemmGrid, T, 0, stream>>>(
          XT, RDEG, B_S0WL, FIN, xin, B_S0WR, FIN, s_b[0], H0, N_NODES, FIN, HID, HID, 1);
    else if (li == 1)
      wmma_gemm_k<16, true, true><<<gemmGrid, T, 0, stream>>>(
          XT, RDEG, B_S1WL, HID, xin, B_S1WR, HID, s_b[1], H1, N_NODES, HID, HID, HID, 1);
    else
      wmma_gemm_k<3, true, true><<<gemmGrid, T, 0, stream>>>(
          XT, RDEG, B_S2WL, HID, xin, B_S2WR, HID, s_b[2], XS, N_NODES, HID, NC, KHP, 0);
  }

  // --- Hypergraph branch: xt = g@w^T; ef = B^-1 * scatter(xt[he_node]->he_edge);
  //     g' = relu(D^-1 * scatter(ef[he_edge]->he_node) + b)
  const float* gin[5] = {x, H0, H1, H0, H1};
  float* gout[5] = {H0, H1, H0, H1, XH};
  const __bf16* hwb[5] = {B_H0, B_H1, B_H2, B_H3, B_H4};
  int hI[5] = {FIN, HID, HID, HID, HID};
  int hO[5] = {HID, HID, HID, HID, NC};
  for (int li = 0; li < 5; ++li) {
    int I = hI[li], O = hO[li];
    int osh = (O == HID) ? 8 : -1;
    int ldo = (li == 4) ? KHP : O;          // XH is stride-64 padded
    if (li < 4)
      wmma_gemm_k<16, false, false><<<gemmGrid, T, 0, stream>>>(
          gin[li], nullptr, hwb[li], I, nullptr, nullptr, 0, nullptr, XT,
          N_NODES, I, O, O, 0);
    else
      wmma_gemm_k<3, false, false><<<gemmGrid, T, 0, stream>>>(
          gin[li], nullptr, hwb[li], I, nullptr, nullptr, 0, nullptr, XT,
          N_NODES, I, O, O, 0);
    fill_zero_k<<<cdiv((long long)M_HE * O, T), T, 0, stream>>>(EF, M_HE * O);
    scatter_k<<<cdiv((long long)N_INC * O, T), T, 0, stream>>>(
        XT, he_node, he_edge, nullptr, EF, N_INC, O, osh, O);
    if (li < 4)
      fill_zero_k<<<cdiv((long long)N_NODES * O, T), T, 0, stream>>>(gout[li], N_NODES * O);
    // (XH already zero-filled with its pad columns)
    scatter_k<<<cdiv((long long)N_INC * O, T), T, 0, stream>>>(
        EF, he_edge, he_node, BINV, gout[li], N_INC, O, osh, ldo);   // fold B^-1 in
    epi_k<<<cdiv((long long)N_NODES * O, T), T, 0, stream>>>(
        gout[li], DINV, h_b[li], N_NODES, O, osh, ldo, li < 4 ? 1 : 0);
  }

  // --- head: logits = XS @ lp_w[:, :40]^T + XH @ lp_w[:, 40:]^T + b (K padded to 64)
  wmma_gemm_k<3, true, false><<<gemmGrid, T, 0, stream>>>(
      XS, nullptr, B_LW1, KHP, XH, B_LW2, KHP, lp_b, out,
      N_NODES, KHP, NC, NC, 0);
  log_softmax_k<<<cdiv(N_NODES, T), T, 0, stream>>>(out, N_NODES, NC);
}
